// OverfitResonanceModel_13202729468290
// MI455X (gfx1250) — compile-verified, matrix operand order checked by reading, not run on hardware
//
#include <hip/hip_runtime.h>
#include <math.h>

// ---- problem sizes ----
#define NE 32
#define S_N 65536
#define FR 128
#define NRES 1024
#define IEXP 8
#define NEXP 4
#define NFILT 64
#define FSAMP 512
#define NDEF 64
#define NNDEF 32
#define NENV 64
#define NDEC 64
#define ENV_SMALL 128
#define ENV_FULL 8192
#define NVERB 16
#define FILTN 8704        // support of filtered = 8192+512-1, padded to x16

typedef __attribute__((ext_vector_type(16))) __bf16 v16bf;
typedef __attribute__((ext_vector_type(2)))  __bf16 v2bf;
typedef __attribute__((ext_vector_type(8)))  float  v8f;
typedef __attribute__((ext_vector_type(4)))  int    i32x4;

// ---- feature probes (safe fallbacks keep the build green) ----
#if defined(__HIP_DEVICE_COMPILE__) && __has_builtin(__builtin_amdgcn_global_load_async_to_lds_b128)
#define HAVE_ASYNC_LDS 1
#else
#define HAVE_ASYNC_LDS 0
#endif

// ---- workspace layout (bytes), 256B-aligned regions ----
constexpr size_t alignup(size_t x) { return (x + 255) & ~(size_t)255; }
constexpr size_t WS_IMP    = 0;
constexpr size_t WS_NF     = alignup(WS_IMP    + (size_t)NE*ENV_FULL*4);
constexpr size_t WS_WDEF   = alignup(WS_NF     + (size_t)NE*NEXP*FSAMP*4);
constexpr size_t WS_NDDEF  = alignup(WS_WDEF   + (size_t)NE*IEXP*FR*4);
constexpr size_t WS_DEC    = alignup(WS_NDDEF  + (size_t)NE*NEXP*FR*4);
constexpr size_t WS_M16    = alignup(WS_DEC    + (size_t)NE*FR*4);
constexpr size_t WS_MIX    = alignup(WS_M16    + (size_t)NE*NVERB*4);
constexpr size_t WS_AMP2   = alignup(WS_MIX    + (size_t)NE*2*4);
constexpr size_t WS_ASOFT  = alignup(WS_AMP2   + (size_t)NE*4);
constexpr size_t WS_RESMIX = alignup(WS_ASOFT  + (size_t)256*NRES*4);   // bf16 256x1024
constexpr size_t WS_ROOMS  = alignup(WS_RESMIX + (size_t)256*NRES*2);   // bf16 NE x S
constexpr size_t WS_DRES   = alignup(WS_ROOMS  + (size_t)NE*S_N*2);     // bf16 NE x S
constexpr size_t WS_FILTBF = alignup(WS_DRES   + (size_t)NE*S_N*2);     // bf16 NE x FILTN
constexpr size_t WS_CONV   = alignup(WS_FILTBF + (size_t)NE*FILTN*2);   // f32 NE x S
constexpr size_t WS_CONVBF = alignup(WS_CONV   + (size_t)NE*S_N*4);     // bf16 NE x S

// ---- helpers ----
__device__ __forceinline__ int imin(int a, int b) { return a < b ? a : b; }

__device__ __forceinline__ unsigned short f2bfb_sw(float f) {
  unsigned u = __builtin_bit_cast(unsigned, f);
  unsigned r = u + 0x7FFFu + ((u >> 16) & 1u);   // round-to-nearest-even
  return (unsigned short)(r >> 16);
}

// packed f32x2 -> bf16x2 : prefer the HW packed convert; otherwise let clang
// fuse build_vector(fptrunc,fptrunc) into v_cvt_pk_bf16_f32 when the target
// has it.
__device__ __forceinline__ unsigned pk_bf16(float lo, float hi) {
#if defined(__HIP_DEVICE_COMPILE__)
#if __has_builtin(__builtin_amdgcn_cvt_pk_bf16_f32)
  v2bf p = __builtin_amdgcn_cvt_pk_bf16_f32(lo, hi);
  return __builtin_bit_cast(unsigned, p);
#else
  v2bf p;
  p[0] = (__bf16)lo;
  p[1] = (__bf16)hi;
  return __builtin_bit_cast(unsigned, p);
#endif
#else
  return (unsigned)f2bfb_sw(lo) | ((unsigned)f2bfb_sw(hi) << 16);
#endif
}

// single f32 -> bf16 bits, via native fptrunc lowering
__device__ __forceinline__ unsigned short f2bfb(float f) {
#if defined(__HIP_DEVICE_COMPILE__)
  __bf16 b = (__bf16)f;
  return __builtin_bit_cast(unsigned short, b);
#else
  return f2bfb_sw(f);
#endif
}

// 16-byte global -> LDS copy (async path tracked by ASYNCcnt when available)
__device__ __forceinline__ void async_cp16(const void* g, void* l) {
#if HAVE_ASYNC_LDS
  __builtin_amdgcn_global_load_async_to_lds_b128(
      (__attribute__((address_space(1))) i32x4*)(g),
      (__attribute__((address_space(3))) i32x4*)(l), 0, 0);
#else
  *(uint4*)l = *(const uint4*)g;
#endif
}
__device__ __forceinline__ void async_cp_wait() {
#if HAVE_ASYNC_LDS
#if __has_builtin(__builtin_amdgcn_s_wait_asynccnt)
  __builtin_amdgcn_s_wait_asynccnt(0);
#endif
#endif
}

__device__ __forceinline__ float rndpm1(unsigned x) {
  x = (x ^ 61u) ^ (x >> 16); x *= 9u; x ^= x >> 4;
  x *= 0x27d4eb2du; x ^= x >> 15;
  return (float)x * (2.0f / 4294967296.0f) - 1.0f;
}

// even K base of bf16 WMMA fragment pair v (v in 0..7), khalf = 0 or 8
__device__ __forceinline__ int kmap(int v, int khalf) {
  return ((v & 4) ? 16 : 0) + khalf + 2 * (v & 3);
}

// ====================================================================
// Kernel 1: all small softmax lookups, per event (one block per event)
// ====================================================================
__global__ __launch_bounds__(128) void k_prepare(
    const float* __restrict__ res_sel, const float* __restrict__ noise_sel,
    const float* __restrict__ env_sel, const float* __restrict__ dec_sel,
    const float* __restrict__ def_sel, const float* __restrict__ noise_def_sel,
    const float* __restrict__ mix_sel, const float* __restrict__ room_sel,
    const float* __restrict__ amp, const float* __restrict__ n_items,
    const float* __restrict__ e_items, const float* __restrict__ d_items,
    const float* __restrict__ warp_items, const float* __restrict__ nwarp_items,
    unsigned char* __restrict__ ws)
{
  const int e = blockIdx.x;
  const int t = threadIdx.x;
  float* imp   = (float*)(ws + WS_IMP)   + (size_t)e * ENV_FULL;
  float* nf    = (float*)(ws + WS_NF)    + (size_t)e * NEXP * FSAMP;
  float* wdef  = (float*)(ws + WS_WDEF)  + (size_t)e * IEXP * FR;
  float* nddef = (float*)(ws + WS_NDDEF) + (size_t)e * NEXP * FR;
  float* dec   = (float*)(ws + WS_DEC)   + (size_t)e * FR;
  float* m16   = (float*)(ws + WS_M16)   + (size_t)e * NVERB;
  float* mixw  = (float*)(ws + WS_MIX)   + (size_t)e * 2;
  float* amp2  = (float*)(ws + WS_AMP2)  + e;
  float* asoft = (float*)(ws + WS_ASOFT) + (size_t)e * IEXP * NRES;

  __shared__ float envS[ENV_SMALL];
  __shared__ float flatS[IEXP * FR];

  // ---- envelope: softmax @ e_items^2 ----
  {
    const float* row = env_sel + e * NENV;
    float mx = -1e30f; for (int k = 0; k < NENV; k++) mx = fmaxf(mx, row[k]);
    float den = 0;     for (int k = 0; k < NENV; k++) den += __expf(row[k] - mx);
    float inv = 1.0f / den;
    if (t < ENV_SMALL) {
      float acc = 0;
      for (int k = 0; k < NENV; k++) {
        float w = __expf(row[k] - mx) * inv;
        float it = e_items[k * ENV_SMALL + t];
        acc += w * it * it;
      }
      envS[t] = acc;
    }
  }
  __syncthreads();
  for (int j = t; j < ENV_FULL; j += 128) {
    float pos = j * ((float)(ENV_SMALL - 1) / (float)(ENV_FULL - 1));
    int lo = (int)pos; int hi = imin(lo + 1, ENV_SMALL - 1); float f = pos - (float)lo;
    float v = envS[lo] * (1.0f - f) + envS[hi] * f;
    imp[j] = v * rndpm1((unsigned)(e * ENV_FULL + j) * 747796405u + 2891336453u);
  }
  // ---- noise filters: softmax @ (n_items * hamming) ----
  {
    const float* rowb = noise_sel + e * NEXP * NFILT;
    float mxs[NEXP], invs[NEXP];
    for (int x = 0; x < NEXP; x++) {
      const float* row = rowb + x * NFILT;
      float mx = -1e30f; for (int k = 0; k < NFILT; k++) mx = fmaxf(mx, row[k]);
      float den = 0;     for (int k = 0; k < NFILT; k++) den += __expf(row[k] - mx);
      mxs[x] = mx; invs[x] = 1.0f / den;
    }
    for (int idx = t; idx < NEXP * FSAMP; idx += 128) {
      int x = idx / FSAMP, j = idx % FSAMP;
      float ham = 0.54f - 0.46f * __cosf(6.2831853071795864f * (float)j / (float)(FSAMP - 1));
      const float* row = rowb + x * NFILT;
      float acc = 0;
      for (int k = 0; k < NFILT; k++)
        acc += __expf(row[k] - mxs[x]) * invs[x] * n_items[k * FSAMP + j];
      nf[idx] = acc * ham;
    }
  }
  // ---- instrument deformations ----
  {
    const float* row = def_sel + e * NDEF;
    float mx = -1e30f; for (int k = 0; k < NDEF; k++) mx = fmaxf(mx, row[k]);
    float den = 0;     for (int k = 0; k < NDEF; k++) den += __expf(row[k] - mx);
    float inv = 1.0f / den;
    for (int j = t; j < IEXP * FR; j += 128) {
      float acc = 0;
      for (int k = 0; k < NDEF; k++)
        acc += __expf(row[k] - mx) * inv * warp_items[k * (IEXP * FR) + j];
      flatS[j] = acc;
    }
  }
  __syncthreads();
  if (t < FR) {
    float mx = -1e30f; for (int i = 0; i < IEXP; i++) mx = fmaxf(mx, flatS[i * FR + t]);
    float den = 0;     for (int i = 0; i < IEXP; i++) den += __expf(flatS[i * FR + t] - mx);
    float inv = 1.0f / den;
    for (int i = 0; i < IEXP; i++) wdef[i * FR + t] = __expf(flatS[i * FR + t] - mx) * inv;
  }
  __syncthreads();
  // ---- noise deformations (reuse flatS) ----
  {
    const float* row = noise_def_sel + e * NNDEF;
    float mx = -1e30f; for (int k = 0; k < NNDEF; k++) mx = fmaxf(mx, row[k]);
    float den = 0;     for (int k = 0; k < NNDEF; k++) den += __expf(row[k] - mx);
    float inv = 1.0f / den;
    for (int j = t; j < NEXP * FR; j += 128) {
      float acc = 0;
      for (int k = 0; k < NNDEF; k++)
        acc += __expf(row[k] - mx) * inv * nwarp_items[k * (NEXP * FR) + j];
      flatS[j] = acc;
    }
  }
  __syncthreads();
  if (t < FR) {
    float mx = -1e30f; for (int x = 0; x < NEXP; x++) mx = fmaxf(mx, flatS[x * FR + t]);
    float den = 0;     for (int x = 0; x < NEXP; x++) den += __expf(flatS[x * FR + t] - mx);
    float inv = 1.0f / den;
    for (int x = 0; x < NEXP; x++) nddef[x * FR + t] = __expf(flatS[x * FR + t] - mx) * inv;
  }
  // ---- decays ----
  {
    const float* row = dec_sel + e * NDEC;
    float mx = -1e30f; for (int k = 0; k < NDEC; k++) mx = fmaxf(mx, row[k]);
    float den = 0;     for (int k = 0; k < NDEC; k++) den += __expf(row[k] - mx);
    float inv = 1.0f / den;
    float d0 = 0;
    for (int k = 0; k < NDEC; k++) {
      float sig = 1.0f / (1.0f + __expf(-d_items[k]));
      d0 += __expf(row[k] - mx) * inv * (0.5f + 0.5f * sig);
    }
    float L = __logf(d0 + 1e-12f);
    if (t < FR) dec[t] = __expf((float)(t + 1) * L);
  }
  // ---- room / mix / amp ----
  if (t < NVERB) {
    const float* row = room_sel + e * NVERB;
    float mx = -1e30f; for (int k = 0; k < NVERB; k++) mx = fmaxf(mx, row[k]);
    float den = 0;     for (int k = 0; k < NVERB; k++) den += __expf(row[k] - mx);
    m16[t] = __expf(row[t] - mx) / den;
  }
  if (t < 2) {
    const float* row = mix_sel + e * 2;
    float mx = fmaxf(row[0], row[1]);
    float den = __expf(row[0] - mx) + __expf(row[1] - mx);
    mixw[t] = __expf(row[t] - mx) / den;
  }
  if (t == 0) { float a = amp[e]; *amp2 = a * a; }
  // ---- softmax(res_sel) rows ----
  for (int i = 0; i < IEXP; i++) {
    const float* row = res_sel + ((size_t)e * IEXP + i) * NRES;
    float mx = -1e30f; for (int k = 0; k < NRES; k++) mx = fmaxf(mx, row[k]);
    float den = 0;     for (int k = 0; k < NRES; k++) den += __expf(row[k] - mx);
    float inv = 1.0f / den;
    for (int k = t; k < NRES; k += 128) asoft[i * NRES + k] = __expf(row[k] - mx) * inv;
  }
}

// ====================================================================
// Kernel 2: resmix = asoft(256x1024) @ r_items(1024x1024) -> bf16
// ====================================================================
__global__ __launch_bounds__(256) void k_resmix(const float* __restrict__ r_items,
                                                unsigned char* __restrict__ ws) {
  const int m = blockIdx.x; const int t = threadIdx.x;
  const float* a = (const float*)(ws + WS_ASOFT) + (size_t)m * NRES;
  unsigned* out = (unsigned*)((unsigned short*)(ws + WS_RESMIX) + (size_t)m * NRES);
  float a0 = 0, a1 = 0, a2 = 0, a3 = 0;
  for (int k = 0; k < NRES; k++) {
    float av = a[k];
    const float* br = r_items + (size_t)k * NRES + 2 * t;
    const float2 p0 = *(const float2*)br;
    const float2 p1 = *(const float2*)(br + 512);
    a0 += av * p0.x; a1 += av * p0.y; a2 += av * p1.x; a3 += av * p1.y;
  }
  out[t] = pk_bf16(a0, a1);
  out[256 + t] = pk_bf16(a2, a3);
}

// ====================================================================
// Kernel 3: rooms[e][s] = sum_v m16[e][v] * verbs[v][s] -> bf16
// ====================================================================
__global__ __launch_bounds__(256) void k_rooms(const float* __restrict__ verbs,
                                               unsigned char* __restrict__ ws) {
  const int e = blockIdx.y;
  const int s0 = blockIdx.x * 1024 + threadIdx.x * 4;
  const float* m16 = (const float*)(ws + WS_M16) + (size_t)e * NVERB;
  float a0 = 0, a1 = 0, a2 = 0, a3 = 0;
  for (int v = 0; v < NVERB; v++) {
    const float4 vv = *(const float4*)(verbs + (size_t)v * S_N + s0);
    float w = m16[v];
    a0 += w * vv.x; a1 += w * vv.y; a2 += w * vv.z; a3 += w * vv.w;
  }
  unsigned short* rm = (unsigned short*)(ws + WS_ROOMS) + (size_t)e * S_N + s0;
  uint2 pk;
  pk.x = pk_bf16(a0, a1);
  pk.y = pk_bf16(a2, a3);
  *(uint2*)rm = pk;
}

// ====================================================================
// Kernel 4: WMMA wavetable GEMM, M=256 (all events*IEXP), K=1024,
// N chunk = 64, fused decay/deform epilogue -> dres bf16.
// waves (256MB) is streamed exactly once across the grid.
// ====================================================================
__global__ __launch_bounds__(256) void k_wavetable(const float* __restrict__ waves,
                                                   unsigned char* __restrict__ ws) {
  const int nb = blockIdx.x * 64;
  const int wv = threadIdx.x >> 5;
  const int lane = threadIdx.x & 31;
  const int lr = lane & 15;
  const int khalf = (lane >> 4) ? 8 : 0;
  const unsigned short* resmix = (const unsigned short*)(ws + WS_RESMIX);

  v8f acc[2][4];
  for (int tt = 0; tt < 2; tt++)
    for (int nt = 0; nt < 4; nt++)
      acc[tt][nt] = (v8f){0, 0, 0, 0, 0, 0, 0, 0};

  const unsigned* r0 = (const unsigned*)(resmix + (size_t)(32 * wv + lr) * NRES);
  const unsigned* r1 = (const unsigned*)(resmix + (size_t)(32 * wv + 16 + lr) * NRES);

#pragma unroll 1
  for (int kb = 0; kb < NRES; kb += 32) {
    union { v16bf v; unsigned u[8]; } A0, A1, B[4];
#pragma unroll
    for (int v = 0; v < 8; v++) {
      int k0 = kb + kmap(v, khalf);
      A0.u[v] = r0[k0 >> 1];
      A1.u[v] = r1[k0 >> 1];
    }
    if (kb + 32 < NRES)   // prefetch next K-slab of the waves stream
      __builtin_prefetch(waves + (size_t)(kb + 32 + khalf) * S_N + nb + lr, 0, 1);
#pragma unroll
    for (int nt = 0; nt < 4; nt++) {
      int n = nb + nt * 16 + lr;
#pragma unroll
      for (int v = 0; v < 8; v++) {
        int k0 = kb + kmap(v, khalf);
        float w0 = waves[(size_t)k0 * S_N + n];
        float w1 = waves[(size_t)(k0 + 1) * S_N + n];
        B[nt].u[v] = pk_bf16(w0, w1);
      }
    }
#pragma unroll
    for (int nt = 0; nt < 4; nt++) {
      acc[0][nt] = __builtin_amdgcn_wmma_f32_16x16x32_bf16(
          false, A0.v, false, B[nt].v, (short)0, acc[0][nt], false, false);
      acc[1][nt] = __builtin_amdgcn_wmma_f32_16x16x32_bf16(
          false, A1.v, false, B[nt].v, (short)0, acc[1][nt], false, false);
    }
  }
  // epilogue: dres[e][s] = dec(s) * sum_i res[e,i,s] * wdef[e,i,s]
  const float* wdefb = (const float*)(ws + WS_WDEF);
  const float* decb  = (const float*)(ws + WS_DEC);
  unsigned short* dres = (unsigned short*)(ws + WS_DRES);
#pragma unroll
  for (int tt = 0; tt < 2; tt++) {
    int e = 4 * wv + 2 * tt + (lane >> 4);
    const float* wd = wdefb + (size_t)e * IEXP * FR;
    const float* dc = decb + (size_t)e * FR;
#pragma unroll
    for (int nt = 0; nt < 4; nt++) {
      int s = nb + nt * 16 + lr;
      float pos = (float)s * ((float)(FR - 1) / (float)(S_N - 1));
      int lo = (int)pos; int hi = imin(lo + 1, FR - 1); float f = pos - (float)lo;
      float sum = 0;
#pragma unroll
      for (int r = 0; r < 8; r++) {
        float wr = wd[r * FR + lo] * (1.0f - f) + wd[r * FR + hi] * f;
        sum += acc[tt][nt][r] * wr;
      }
      float dv = dc[lo] * (1.0f - f) + dc[hi] * f;
      dres[(size_t)e * S_N + s] = f2bfb(sum * dv);
    }
  }
}

// ====================================================================
// Kernel 5: filtered[e][n] = sum_x nd_x(n) * (imp * nf_x)[n]  -> bf16
// ====================================================================
__global__ __launch_bounds__(256) void k_filt(unsigned char* __restrict__ ws) {
  const int e = blockIdx.y;
  const int nb = blockIdx.x * 256;
  const int t = threadIdx.x;
  const int n = nb + t;
  __shared__ float nfS[NEXP * FSAMP];
  __shared__ float impS[FSAMP + 256];
  const float* nf  = (const float*)(ws + WS_NF)  + (size_t)e * NEXP * FSAMP;
  const float* imp = (const float*)(ws + WS_IMP) + (size_t)e * ENV_FULL;
  for (int j = t; j < NEXP * FSAMP; j += 256) nfS[j] = nf[j];
  for (int j = t; j < FSAMP + 256; j += 256) {
    int gi = nb - (FSAMP - 1) + j;
    impS[j] = (gi >= 0 && gi < ENV_FULL) ? imp[gi] : 0.0f;
  }
  __syncthreads();
  float acc[NEXP] = {0, 0, 0, 0};
  for (int j = 0; j < FSAMP; j++) {
    float iv = impS[t + (FSAMP - 1) - j];
#pragma unroll
    for (int x = 0; x < NEXP; x++) acc[x] += nfS[x * FSAMP + j] * iv;
  }
  const float* nd = (const float*)(ws + WS_NDDEF) + (size_t)e * NEXP * FR;
  float pos = (float)n * ((float)(FR - 1) / (float)(S_N - 1));
  int lo = (int)pos; int hi = imin(lo + 1, FR - 1); float f = pos - (float)lo;
  float outv = 0;
#pragma unroll
  for (int x = 0; x < NEXP; x++) {
    float ndv = nd[x * FR + lo] * (1.0f - f) + nd[x * FR + hi] * f;
    outv += ndv * acc[x];
  }
  ((unsigned short*)(ws + WS_FILTBF))[(size_t)e * FILTN + n] = f2bfb(outv);
}

// ====================================================================
// Kernel 6: conv = filtered (x) dres, banded block-Toeplitz WMMA GEMM.
// Each wave: 1 shared A_s fragment x 4 N-tiles (64 q) per k-step.
// F staged once into LDS as bf16 (async copy when available).
// ====================================================================
__global__ __launch_bounds__(256) void k_conv(unsigned char* __restrict__ ws) {
  const int e = blockIdx.y;
  const int wv = threadIdx.x >> 5, lane = threadIdx.x & 31;
  const int lr = lane & 15, hb = lane >> 4;
  const int khalf = hb ? 8 : 0;
  const int qb0 = blockIdx.x * 512 + wv * 64;       // 4 tiles of 16 q per wave
  __shared__ unsigned short Fb[FILTN];
  {
    const uint4* src = (const uint4*)((const unsigned short*)(ws + WS_FILTBF) +
                                      (size_t)e * FILTN);
    uint4* dst = (uint4*)Fb;
    for (int j = threadIdx.x; j < FILTN / 8; j += 256) async_cp16(src + j, dst + j);
    async_cp_wait();
  }
  __syncthreads();
  const unsigned short* dres = (const unsigned short*)(ws + WS_DRES) + (size_t)e * S_N;
  v8f acc[4];
#pragma unroll
  for (int j = 0; j < 4; j++) acc[j] = (v8f){0, 0, 0, 0, 0, 0, 0, 0};
  const int p = lr;
  const int smax = imin(FILTN / 16 - 1, qb0 + 62);
#pragma unroll 1
  for (int s = 0; s <= smax; s++) {
    union { v16bf v; unsigned u[8]; } A, B[4];
#pragma unroll
    for (int v = 0; v < 8; v++) {
      int t0 = kmap(v, khalf);
      int d0 = 16 + p - t0, d1 = d0 - 1;
      bool i0 = (d0 >= 0 && d0 < 16), i1 = (d1 >= 0 && d1 < 16);
      unsigned short b0 = Fb[16 * s + (i0 ? d0 : 0)];
      unsigned short b1 = Fb[16 * s + (i1 ? d1 : 0)];
      A.u[v] = (unsigned)(i0 ? b0 : 0) | ((unsigned)(i1 ? b1 : 0) << 16);
#pragma unroll
      for (int j = 0; j < 4; j++) {
        int idx = 16 * (qb0 + 16 * j + lr - s - 1) + t0;
        B[j].u[v] = (idx >= 0) ? *(const unsigned*)(dres + idx) : 0u;
      }
    }
#pragma unroll
    for (int j = 0; j < 4; j++)
      acc[j] = __builtin_amdgcn_wmma_f32_16x16x32_bf16(
          false, A.v, false, B[j].v, (short)0, acc[j], false, false);
  }
  float* conv = (float*)(ws + WS_CONV) + (size_t)e * S_N;
  unsigned short* convb = (unsigned short*)(ws + WS_CONVBF) + (size_t)e * S_N;
#pragma unroll
  for (int j = 0; j < 4; j++) {
    int q = qb0 + 16 * j + lr;
#pragma unroll
    for (int r = 0; r < 8; r++) {
      int n = 16 * q + r + 8 * hb;
      conv[n] = acc[j][r];
      convb[n] = f2bfb(acc[j][r]);
    }
  }
}

// ====================================================================
// Kernel 7: wet = conv (x) rooms, full-length Toeplitz WMMA.
// F (conv, bf16) double-buffered through LDS with async copies;
// fused final mix: out = (m0*conv + m1*wet) * amp^2
// ====================================================================
#define RCHUNK 256
__global__ __launch_bounds__(256) void k_reverb(unsigned char* __restrict__ ws,
                                                float* __restrict__ out) {
  const int e = blockIdx.y;
  const int wv = threadIdx.x >> 5, lane = threadIdx.x & 31;
  const int lr = lane & 15, hb = lane >> 4;
  const int khalf = hb ? 8 : 0;
  const int qb0 = blockIdx.x * 512 + wv * 64;
  __shared__ unsigned short Fb[2][RCHUNK * 16];
  const float* conv = (const float*)(ws + WS_CONV) + (size_t)e * S_N;
  const unsigned short* convb = (const unsigned short*)(ws + WS_CONVBF) + (size_t)e * S_N;
  const unsigned short* rooms = (const unsigned short*)(ws + WS_ROOMS) + (size_t)e * S_N;

  auto stage = [&](int buf, int sc) {
    const uint4* src = (const uint4*)(convb + (size_t)sc * RCHUNK * 16);
    uint4* dst = (uint4*)Fb[buf];
    for (int j = threadIdx.x; j < RCHUNK * 16 / 8; j += 256) async_cp16(src + j, dst + j);
  };

  v8f acc[4];
#pragma unroll
  for (int j = 0; j < 4; j++) acc[j] = (v8f){0, 0, 0, 0, 0, 0, 0, 0};
  const int p = lr;
  const int smax = imin(S_N / 16 - 1, qb0 + 62);
  const int NCH = S_N / 16 / RCHUNK;   // 16 chunks

  stage(0, 0);
#pragma unroll 1
  for (int sc = 0; sc < NCH; ++sc) {
    async_cp_wait();
    __syncthreads();
    if (sc + 1 < NCH) stage((sc + 1) & 1, sc + 1);
    const unsigned short* F = Fb[sc & 1];
    const int s0 = sc * RCHUNK, s1 = imin(smax, s0 + RCHUNK - 1);
#pragma unroll 1
    for (int s = s0; s <= s1; s++) {
      union { v16bf v; unsigned u[8]; } A, B[4];
      const int mloc = 16 * (s - s0);
#pragma unroll
      for (int v = 0; v < 8; v++) {
        int t0 = kmap(v, khalf);
        int d0 = 16 + p - t0, d1 = d0 - 1;
        bool i0 = (d0 >= 0 && d0 < 16), i1 = (d1 >= 0 && d1 < 16);
        unsigned short b0 = F[mloc + (i0 ? d0 : 0)];
        unsigned short b1 = F[mloc + (i1 ? d1 : 0)];
        A.u[v] = (unsigned)(i0 ? b0 : 0) | ((unsigned)(i1 ? b1 : 0) << 16);
#pragma unroll
        for (int j = 0; j < 4; j++) {
          int idx = 16 * (qb0 + 16 * j + lr - s - 1) + t0;
          B[j].u[v] = (idx >= 0) ? *(const unsigned*)(rooms + idx) : 0u;
        }
      }
#pragma unroll
      for (int j = 0; j < 4; j++)
        acc[j] = __builtin_amdgcn_wmma_f32_16x16x32_bf16(
            false, A.v, false, B[j].v, (short)0, acc[j], false, false);
    }
    __syncthreads();
  }
  const float* mixw = (const float*)(ws + WS_MIX) + (size_t)e * 2;
  const float a2 = ((const float*)(ws + WS_AMP2))[e];
  const float m0 = mixw[0], m1 = mixw[1];
#pragma unroll
  for (int j = 0; j < 4; j++) {
    int q = qb0 + 16 * j + lr;
#pragma unroll
    for (int r = 0; r < 8; r++) {
      int n = 16 * q + r + 8 * hb;
      float cv = conv[n];
      out[(size_t)e * S_N + n] = (m0 * cv + m1 * acc[j][r]) * a2;
    }
  }
}

// ====================================================================
extern "C" void kernel_launch(void* const* d_in, const int* in_sizes, int n_in,
                              void* d_out, int out_size, void* d_ws, size_t ws_size,
                              hipStream_t stream) {
  (void)in_sizes; (void)n_in; (void)out_size; (void)ws_size;
  const float* res_sel       = (const float*)d_in[0];
  const float* noise_sel     = (const float*)d_in[1];
  const float* env_sel       = (const float*)d_in[2];
  const float* dec_sel       = (const float*)d_in[3];
  const float* def_sel       = (const float*)d_in[4];
  const float* noise_def_sel = (const float*)d_in[5];
  const float* mix_sel       = (const float*)d_in[6];
  const float* room_sel      = (const float*)d_in[7];
  const float* amp           = (const float*)d_in[8];
  const float* r_items       = (const float*)d_in[9];
  const float* n_items       = (const float*)d_in[10];
  const float* e_items       = (const float*)d_in[11];
  const float* d_items       = (const float*)d_in[12];
  const float* warp_items    = (const float*)d_in[13];
  const float* nwarp_items   = (const float*)d_in[14];
  const float* verbs         = (const float*)d_in[15];
  const float* waves         = (const float*)d_in[16];
  unsigned char* ws = (unsigned char*)d_ws;
  float* out = (float*)d_out;

  k_prepare<<<NE, 128, 0, stream>>>(res_sel, noise_sel, env_sel, dec_sel, def_sel,
                                    noise_def_sel, mix_sel, room_sel, amp,
                                    n_items, e_items, d_items, warp_items,
                                    nwarp_items, ws);
  k_resmix<<<256, 256, 0, stream>>>(r_items, ws);
  k_rooms<<<dim3(S_N / 1024, NE), 256, 0, stream>>>(verbs, ws);
  k_wavetable<<<dim3(S_N / 64), 256, 0, stream>>>(waves, ws);
  k_filt<<<dim3(FILTN / 256, NE), 256, 0, stream>>>(ws);
  k_conv<<<dim3(S_N / 16 / 512, NE), 256, 0, stream>>>(ws);
  k_reverb<<<dim3(S_N / 16 / 512, NE), 256, 0, stream>>>(ws, out);
}
